// PruningDecoder_83434034692202
// MI455X (gfx1250) — matrix operand order repaired and checked by reading, not compiled, not run on hardware
//
#include <hip/hip_runtime.h>
#include <hip/hip_bf16.h>
#include <math.h>

// CDNA5 / gfx1250: wave32, WMMA 16x16x32 bf16 with f32 accumulate.
typedef __attribute__((ext_vector_type(16))) __bf16 v16bf;
typedef __attribute__((ext_vector_type(8)))  float  v8f;

#define DD  128   // embedding dim (D == H*DK)
#define HH_ 256   // hyper hidden

// ---------------------------------------------------------------------------
// K1a: hypernetwork (tiny): pref[2] -> 256 -> 256 -> mid[4]. One block.
// ---------------------------------------------------------------------------
__global__ void hyper_kernel(const float* __restrict__ pref,
                             const float* __restrict__ fc1_w, const float* __restrict__ fc1_b,
                             const float* __restrict__ fc2_w, const float* __restrict__ fc2_b,
                             const float* __restrict__ fc3_w, const float* __restrict__ fc3_b,
                             float* __restrict__ mid_out) {
  __shared__ float h1[HH_];
  __shared__ float h2[HH_];
  const int t = threadIdx.x;
  const float p0 = pref[0], p1 = pref[1];
  h1[t] = fc1_w[t * 2 + 0] * p0 + fc1_w[t * 2 + 1] * p1 + fc1_b[t];
  __syncthreads();
  float s = fc2_b[t];
  for (int k = 0; k < HH_; ++k) s += fc2_w[t * HH_ + k] * h1[k];
  h2[t] = s;
  __syncthreads();
  if (t < 4) {
    float m = fc3_b[t];
    for (int k = 0; k < HH_; ++k) m += fc3_w[t * HH_ + k] * h2[k];
    mid_out[t] = m;
  }
}

// ---------------------------------------------------------------------------
// K1b: M = Wq^T Wk (128x128), stored TRANSPOSED (Mt[c][r] = M[r][c]) as bf16
// hi/lo split: score kernel reads contiguous 32B B-fragments and recovers
// near-f32 precision via 3-product split accumulation.
//   Wq[d,r] = mid0*wq_w[d*128+r,0] + mid1*wq_w[d*128+r,1]   (same for Wk/mid2,3)
// ---------------------------------------------------------------------------
__global__ void build_m_kernel(const float* __restrict__ mid,
                               const float* __restrict__ wq_w, const float* __restrict__ wk_w,
                               unsigned short* __restrict__ mt_hi_u,
                               unsigned short* __restrict__ mt_lo_u) {
  const int r = blockIdx.x;   // 0..127
  const int c = threadIdx.x;  // 0..127
  const float m0 = mid[0], m1 = mid[1], m2 = mid[2], m3 = mid[3];
  float s = 0.f;
  for (int d = 0; d < DD; ++d) {
    const float* q = wq_w + (size_t)(d * DD + r) * 2;
    const float* k = wk_w + (size_t)(d * DD + c) * 2;
    const float wq = m0 * q[0] + m1 * q[1];
    const float wk = m2 * k[0] + m3 * k[1];
    s += wq * wk;
  }
  __bf16* mt_hi = reinterpret_cast<__bf16*>(mt_hi_u);
  __bf16* mt_lo = reinterpret_cast<__bf16*>(mt_lo_u);
  const __bf16 h = (__bf16)s;
  mt_hi[c * DD + r] = h;
  mt_lo[c * DD + r] = (__bf16)(s - (float)h);
}

// ---------------------------------------------------------------------------
// K2: segment sums + counts via global f32 atomics. 32 threads per edge,
// each adds a float4 chunk of the 128-wide row (coalesced loads).
// ---------------------------------------------------------------------------
__global__ void segsum_kernel(const float* __restrict__ emb, const int* __restrict__ seg,
                              float* __restrict__ gsum, float* __restrict__ counts, int E) {
  const long long idx = (long long)blockIdx.x * blockDim.x + threadIdx.x;
  const long long e = idx >> 5;
  if (e >= E) return;
  const int c4 = ((int)idx & 31) * 4;
  const int s = seg[e];
  const float4 v = *(const float4*)(emb + (size_t)e * DD + c4);
  float* dst = gsum + (size_t)s * DD + c4;
  atomicAdd(dst + 0, v.x);
  atomicAdd(dst + 1, v.y);
  atomicAdd(dst + 2, v.z);
  atomicAdd(dst + 3, v.w);
  if (((int)idx & 31) == 0) atomicAdd(counts + s, 1.0f);
}

// ---------------------------------------------------------------------------
// A-fragment builder: 16x32 bf16 A fragment, contiguous k order (same as B):
//   lanes 0-15 row M=lane, elems j=0..15 -> K=j (+16 for hi lanes)
// Loads the lane's 16 f32 from the (scaled) q row and splits hi/lo bf16.
// ---------------------------------------------------------------------------
__device__ __forceinline__ void make_a_frag(const float* __restrict__ row, float scale,
                                            int kb, int koff, v16bf& hi, v16bf& lo) {
  const float4* p0 = (const float4*)(row + kb * 32 + koff);
  const float4* p1 = (const float4*)(row + kb * 32 + 8 + koff);
  const float4 a = p0[0], b = p0[1], c = p1[0], d = p1[1];
  float buf[16] = {a.x, a.y, a.z, a.w, b.x, b.y, b.z, b.w,
                   c.x, c.y, c.z, c.w, d.x, d.y, d.z, d.w};
#pragma unroll
  for (int j = 0; j < 16; ++j) {
    const float f = buf[j] * scale;
    const __bf16 h = (__bf16)f;
    hi[j] = h;
    lo[j] = (__bf16)(f - (float)h);
  }
}

// ---------------------------------------------------------------------------
// K3: fused score kernel. One wave = 16 edges.
//   T = Q_tile(16x128) @ M(128x128) via v_wmma_f32_16x16x32_bf16 (split bf16),
//   score_e = rowdot(T, Emb_tile)/(8*sqrt(128)) - d_e/sqrt(2),
//   ex = exp(10*tanh(score)); atomicAdd into group denom.
// Segment-max is skipped: tanh clips |score|<=10 so exp() is safe in f32.
// All inner-loop addresses are base-pointer + compile-time immediate offsets
// so loads cluster into clauses and hide under the WMMA chain.
// ---------------------------------------------------------------------------
__global__ void __launch_bounds__(256)
score_kernel(const float* __restrict__ gsum, const float* __restrict__ counts,
             const int* __restrict__ seg, const float* __restrict__ emb,
             const unsigned short* __restrict__ mth_u, const unsigned short* __restrict__ mtl_u,
             const float* __restrict__ dists, const float* __restrict__ pref,
             float* __restrict__ exv, float* __restrict__ denom, int E) {
  const int lane = threadIdx.x & 31;
  const int wid  = threadIdx.x >> 5;
  const long long tile = (long long)blockIdx.x * 8 + wid;
  const int e0 = (int)(tile * 16);
  if (e0 >= E) return;                      // wave-uniform: EXEC stays all-1 for WMMA

  const int col16 = lane & 15;
  const int hi16  = lane >> 4;              // 0 for lanes 0-15, 1 for 16-31
  const int akoff = hi16 * 16;              // A-fragment K offset
  const int bkoff = hi16 * 16;              // B-fragment K offset

  // Gather this lane's q row: group mean = group_sum * 1/max(count,1)
  int erow = e0 + col16;
  if (erow > E - 1) erow = E - 1;
  const int sg = seg[erow];
  const float scale = 1.0f / fmaxf(counts[sg], 1.0f);
  const float* qrow = gsum + (size_t)sg * DD;

  v16bf Ah[4], Al[4];
#pragma unroll
  for (int kb = 0; kb < 4; ++kb) make_a_frag(qrow, scale, kb, akoff, Ah[kb], Al[kb]);

  // M base pointers with the lane-variant part folded in; all further
  // indexing is unrolled-constant -> immediate IOFFSET on the loads.
  const __bf16* Mhb = reinterpret_cast<const __bf16*>(mth_u) + (size_t)col16 * DD + bkoff;
  const __bf16* Mlb = reinterpret_cast<const __bf16*>(mtl_u) + (size_t)col16 * DD + bkoff;

  // Clamped emb row pointers (one per accumulator slot), col16 folded in.
  const int rbase = e0 + hi16 * 8;
  const float* ep[8];
#pragma unroll
  for (int v = 0; v < 8; ++v) {
    int e = rbase + v;
    if (e > E - 1) e = E - 1;
    ep[v] = emb + (size_t)e * DD + col16;
  }

  float partial[8];
#pragma unroll
  for (int v = 0; v < 8; ++v) partial[v] = 0.f;

#pragma unroll
  for (int ct = 0; ct < 8; ++ct) {
    // Issue the 8 dot-product operands first so they overlap the WMMAs.
    float ev[8];
#pragma unroll
    for (int v = 0; v < 8; ++v) ev[v] = ep[v][ct * 16];

    v8f acc = {};
#pragma unroll
    for (int kb = 0; kb < 4; ++kb) {
      const v16bf bh = *(const v16bf*)(Mhb + ct * 16 * DD + kb * 32);
      const v16bf bl = *(const v16bf*)(Mlb + ct * 16 * DD + kb * 32);
      acc = __builtin_amdgcn_wmma_f32_16x16x32_bf16(false, Ah[kb], false, bh, (short)0, acc, false, false);
      acc = __builtin_amdgcn_wmma_f32_16x16x32_bf16(false, Ah[kb], false, bl, (short)0, acc, false, false);
      acc = __builtin_amdgcn_wmma_f32_16x16x32_bf16(false, Al[kb], false, bh, (short)0, acc, false, false);
    }
    // D layout: elem v -> row M = v + 8*hi16, col N = col16.
#pragma unroll
    for (int v = 0; v < 8; ++v) partial[v] += acc[v] * ev[v];
  }

  // Reduce over the 16 columns held by each half-wave.
#pragma unroll
  for (int v = 0; v < 8; ++v) {
    float p = partial[v];
    p += __shfl_xor(p, 1, 32);
    p += __shfl_xor(p, 2, 32);
    p += __shfl_xor(p, 4, 32);
    p += __shfl_xor(p, 8, 32);
    partial[v] = p;
  }

  if (col16 == 0) {
    const float p0 = pref[0], p1 = pref[1];
    const float invHD = 0.011048543456039806f;   // 1/(8*sqrt(128))
    const float invS2 = 0.70710678118654752f;    // 1/sqrt(2)
#pragma unroll
    for (int v = 0; v < 8; ++v) {
      const int e = rbase + v;
      if (e < E) {
        const float d   = p0 * dists[(size_t)e * 2] + p1 * dists[(size_t)e * 2 + 1];
        const float raw = partial[v] * invHD - d * invS2;
        const float sc  = 10.0f * tanhf(raw);
        const float ex  = expf(sc);               // |sc| <= 10 -> safe
        exv[e] = ex;
        atomicAdd(denom + seg[e], ex);
      }
    }
  }
}

// ---------------------------------------------------------------------------
// K4: probs = ex / denom[seg]
// ---------------------------------------------------------------------------
__global__ void norm_kernel(const float* __restrict__ exv, const float* __restrict__ denom,
                            const int* __restrict__ seg, float* __restrict__ out, int E) {
  const int e = blockIdx.x * blockDim.x + threadIdx.x;
  if (e >= E) return;
  out[e] = exv[e] / denom[seg[e]];
}

// ---------------------------------------------------------------------------
extern "C" void kernel_launch(void* const* d_in, const int* in_sizes, int n_in,
                              void* d_out, int out_size, void* d_ws, size_t ws_size,
                              hipStream_t stream) {
  const float* pref         = (const float*)d_in[0];
  const float* dists        = (const float*)d_in[1];
  const float* edge_emb     = (const float*)d_in[2];
  const int*   shared_nodes = (const int*)d_in[3];
  const float* fc1_w = (const float*)d_in[4];
  const float* fc1_b = (const float*)d_in[5];
  const float* fc2_w = (const float*)d_in[6];
  const float* fc2_b = (const float*)d_in[7];
  const float* fc3_w = (const float*)d_in[8];
  const float* fc3_b = (const float*)d_in[9];
  const float* wq_w  = (const float*)d_in[10];
  const float* wk_w  = (const float*)d_in[11];

  const int E = in_sizes[3];
  if (E <= 0) return;

  char* ws = (char*)d_ws;
  size_t off = 0;
  auto alloc = [&](size_t bytes) {
    size_t o = off;
    off += (bytes + 255) & ~(size_t)255;
    return o;
  };
  const size_t o_mth = alloc((size_t)DD * DD * sizeof(unsigned short));
  const size_t o_mtl = alloc((size_t)DD * DD * sizeof(unsigned short));
  const size_t o_mid = alloc(4 * sizeof(float));
  const size_t o_ex  = alloc((size_t)E * sizeof(float));
  const size_t o_cnt = alloc((size_t)E * sizeof(float));       // zeroed region start
  const size_t o_den = alloc((size_t)E * sizeof(float));
  const size_t o_gs  = alloc((size_t)E * DD * sizeof(float));
  (void)o_gs;

  unsigned short* mth = (unsigned short*)(ws + o_mth);
  unsigned short* mtl = (unsigned short*)(ws + o_mtl);
  float* midp  = (float*)(ws + o_mid);
  float* exv   = (float*)(ws + o_ex);
  float* cnts  = (float*)(ws + o_cnt);
  float* denom = (float*)(ws + o_den);
  float* gsum  = (float*)(ws + o_gs);

  // Zero counts + denom + group_sums (contiguous tail of ws layout).
  hipMemsetAsync(ws + o_cnt, 0, off - o_cnt, stream);

  hyper_kernel<<<1, HH_, 0, stream>>>(pref, fc1_w, fc1_b, fc2_w, fc2_b, fc3_w, fc3_b, midp);
  build_m_kernel<<<DD, DD, 0, stream>>>(midp, wq_w, wk_w, mth, mtl);

  {
    const long long threads = (long long)E * 32;
    const unsigned blocks = (unsigned)((threads + 255) / 256);
    segsum_kernel<<<blocks, 256, 0, stream>>>(edge_emb, shared_nodes, gsum, cnts, E);
  }

  {
    const unsigned blocks = (unsigned)((E + 127) / 128);  // 8 waves x 16 edges per block
    score_kernel<<<blocks, 256, 0, stream>>>(gsum, cnts, shared_nodes, edge_emb,
                                             mth, mtl, dists, pref, exv, denom, E);
  }

  norm_kernel<<<(E + 255) / 256, 256, 0, stream>>>(exv, denom, shared_nodes, (float*)d_out, E);
}
